// Attention_5927054868749
// MI455X (gfx1250) — matrix operand order repaired and checked by reading, not compiled, not run on hardware
//
#include <hip/hip_runtime.h>
#include <hip/hip_bf16.h>

// ---------------------------------------------------------------------------
// SAGAN-style self-attention for MI455X (gfx1250, wave32, WMMA f16->f32).
//
//   0) pack_w:      pre-convert/pre-swizzle all weights into the exact WMMA
//                   A-operand lane layout (f16).
//   1) proj_fused:  one pass over x (64 MB read once) producing
//        thetaT (B,4096,32) f16, phiF (B,4096,32) f16, gF (B,4096,128) f16
//   2) pool2_phi:   phiP (B,1024,32)   key-major (scores B-operand layout)
//      pool2_g:     gT   (B,128,1024)  d-major  (PV B-operand layout)
//   3) attn:        two-pass softmax; S tile (16q x 1024k f16) in LDS per
//                   wave; g blocks staged into LDS with double-buffered
//                   async global->LDS copies distributed over all 8 waves.
//   4) final_conv:  out = sigma * (w_final @ ag^T) + x   (f32, overwrites)
// ---------------------------------------------------------------------------

typedef __attribute__((ext_vector_type(16))) _Float16 v16h;
typedef __attribute__((ext_vector_type(8)))  float    v8f;

// A-operand element e (0..15) of half-wave group g -> K index (16x16x32 f16)
__device__ __forceinline__ int ka_idx(int e, int g) {
  return e + 16 * g;
}
// B-operand element e of group g -> K index
__device__ __forceinline__ int kb_idx(int e, int g) { return e + 16 * g; }

__device__ __forceinline__ v8f zero8() {
  v8f z;
#pragma unroll
  for (int i = 0; i < 8; ++i) z[i] = 0.0f;
  return z;
}

// ---------------------------------------------------------------------------
// Kernel 0: weight pre-pack.  src: w[O][K] f32 row-major.
// dst[t][k][lane][e] = (f16) w[t*16 + (lane&15)][k*32 + ka_idx(e, lane>>4)]
// ---------------------------------------------------------------------------
__global__ void pack_w_kernel(const float* __restrict__ w,
                              _Float16* __restrict__ dst, int ntiles,
                              int nchunks) {
  int idx = blockIdx.x * blockDim.x + threadIdx.x;
  int total = ntiles * nchunks * 32 * 16;
  if (idx >= total) return;
  int e    = idx & 15;
  int lane = (idx >> 4) & 31;
  int k    = (idx >> 9) % nchunks;
  int t    = idx / (nchunks * 512);
  int K    = nchunks * 32;
  int row  = t * 16 + (lane & 15);
  int col  = k * 32 + ka_idx(e, lane >> 4);
  dst[idx] = (_Float16)w[(size_t)row * K + col];
}

// ---------------------------------------------------------------------------
// Kernel 1: fused projections (A-operands pre-packed, one v16h load each).
// ---------------------------------------------------------------------------
__global__ void __launch_bounds__(256) proj_fused_kernel(
    const float* __restrict__ x, const _Float16* __restrict__ wtP,
    const _Float16* __restrict__ wpP, const _Float16* __restrict__ wgP,
    _Float16* __restrict__ thetaT, _Float16* __restrict__ phiF,
    _Float16* __restrict__ gF) {
  const int b = blockIdx.y;
  const int lane = threadIdx.x & 31;
  const int wave = threadIdx.x >> 5;
  const int nbase = blockIdx.x * 128 + wave * 16;
  const int mrow = lane & 15;
  const int grp  = lane >> 4;

  const float* xb = x + (size_t)b * 256 * 4096 + (nbase + mrow);

  v8f accT[2], accP[2], accG[8];
#pragma unroll
  for (int t = 0; t < 2; ++t) { accT[t] = zero8(); accP[t] = zero8(); }
#pragma unroll
  for (int t = 0; t < 8; ++t) accG[t] = zero8();

  auto ldp = [&](const _Float16* pk, int t, int k) {
    return *(const v16h*)(pk + (((size_t)t * 8 + k) * 32 + lane) * 16);
  };

  for (int k = 0; k < 8; ++k) {
    const int kc = k * 32;
    v16h bop;  // B = x chunk (32 channels x 16 spatial)
#pragma unroll
    for (int e = 0; e < 16; ++e)
      bop[e] = (_Float16)xb[(size_t)(kc + kb_idx(e, grp)) * 4096];

#pragma unroll
    for (int t = 0; t < 2; ++t)
      accT[t] = __builtin_amdgcn_wmma_f32_16x16x32_f16(
          false, ldp(wtP, t, k), false, bop, (short)0, accT[t], false, false);
#pragma unroll
    for (int t = 0; t < 2; ++t)
      accP[t] = __builtin_amdgcn_wmma_f32_16x16x32_f16(
          false, ldp(wpP, t, k), false, bop, (short)0, accP[t], false, false);
#pragma unroll
    for (int t = 0; t < 8; ++t)
      accG[t] = __builtin_amdgcn_wmma_f32_16x16x32_f16(
          false, ldp(wgP, t, k), false, bop, (short)0, accG[t], false, false);
  }

  const size_t nrow = (size_t)b * 4096 + nbase + mrow;
  _Float16* yt = thetaT + nrow * 32;
  _Float16* yp = phiF   + nrow * 32;
  _Float16* yg = gF     + nrow * 128;
#pragma unroll
  for (int t = 0; t < 2; ++t)
#pragma unroll
    for (int r = 0; r < 8; ++r) {
      yt[t * 16 + r + 8 * grp] = (_Float16)accT[t][r];
      yp[t * 16 + r + 8 * grp] = (_Float16)accP[t][r];
    }
#pragma unroll
  for (int t = 0; t < 8; ++t)
#pragma unroll
    for (int r = 0; r < 8; ++r)
      yg[t * 16 + r + 8 * grp] = (_Float16)accG[t][r];
}

// ---------------------------------------------------------------------------
// Kernel 2a: 2x2 maxpool phi: (B,4096,32) n-major -> (B,1024,32) key-major.
// ---------------------------------------------------------------------------
__global__ void pool2_phi_kernel(const _Float16* __restrict__ yF,
                                 _Float16* __restrict__ yP, int total) {
  int idx = blockIdx.x * blockDim.x + threadIdx.x;
  if (idx >= total) return;
  int o = idx & 31;
  int m = (idx >> 5) & 1023;
  int b = idx >> 15;
  int h2 = m >> 5, w2 = m & 31;
  int n0 = (h2 * 2) * 64 + w2 * 2;
  const _Float16* base = yF + (size_t)b * 4096 * 32;
  float v0 = (float)base[(size_t)(n0) * 32 + o];
  float v1 = (float)base[(size_t)(n0 + 1) * 32 + o];
  float v2 = (float)base[(size_t)(n0 + 64) * 32 + o];
  float v3 = (float)base[(size_t)(n0 + 65) * 32 + o];
  yP[(size_t)(b * 1024 + m) * 32 + o] =
      (_Float16)fmaxf(fmaxf(v0, v1), fmaxf(v2, v3));
}

// ---------------------------------------------------------------------------
// Kernel 2b: 2x2 maxpool g: (B,4096,128) n-major -> (B,128,1024) d-major.
// ---------------------------------------------------------------------------
__global__ void pool2_g_kernel(const _Float16* __restrict__ yF,
                               _Float16* __restrict__ gT, int total) {
  int idx = blockIdx.x * blockDim.x + threadIdx.x;
  if (idx >= total) return;
  int d = idx & 127;
  int m = (idx >> 7) & 1023;
  int b = idx >> 17;
  int h2 = m >> 5, w2 = m & 31;
  int n0 = (h2 * 2) * 64 + w2 * 2;
  const _Float16* base = yF + (size_t)b * 4096 * 128;
  float v0 = (float)base[(size_t)(n0) * 128 + d];
  float v1 = (float)base[(size_t)(n0 + 1) * 128 + d];
  float v2 = (float)base[(size_t)(n0 + 64) * 128 + d];
  float v3 = (float)base[(size_t)(n0 + 65) * 128 + d];
  gT[((size_t)b * 128 + d) * 1024 + m] =
      (_Float16)fmaxf(fmaxf(v0, v1), fmaxf(v2, v3));
}

// ---------------------------------------------------------------------------
// Kernel 3: two-pass softmax attention with async-LDS g staging.
// LDS: 8 x (16 x 1032h) S tiles (258 KB) + row maxima + 2 x (128 x 40h)
// double-buffered g tile (padded rows: conflict-free reads) ~= 278.5 KB.
// Async copies are distributed: each wave issues 2 of the 16 b128-per-lane
// copies and waits on its own ASYNCcnt before the block barrier.
// ---------------------------------------------------------------------------
__global__ void __launch_bounds__(256) attn_kernel(
    const _Float16* __restrict__ thetaT, const _Float16* __restrict__ phiP,
    const _Float16* __restrict__ gT, _Float16* __restrict__ ag) {
  extern __shared__ char smem[];
  constexpr int SROW = 1032;  // padded S row stride (halves)
  constexpr int GROW = 40;    // padded g row stride (halves), 80B (16B-mult)
  const size_t S_BYTES = (size_t)8 * 16 * SROW * 2;    // 264192
  const size_t R_BYTES = 8 * 16 * 4;                   // 512
  const int b = blockIdx.y;
  const int lane = threadIdx.x & 31;
  const int wave = threadIdx.x >> 5;
  const int qbase = blockIdx.x * 128 + wave * 16;
  const int mrow = lane & 15;
  const int grp  = lane >> 4;

  _Float16* Sw   = (_Float16*)smem + (size_t)wave * 16 * SROW;
  float*    Rmax = (float*)(smem + S_BYTES) + wave * 16;
  _Float16* Gbuf = (_Float16*)(smem + S_BYTES + R_BYTES);  // [2][128][GROW]

  // theta A-operand: row = query, K = 32 channels (contiguous f16 row)
  v16h thA;
  {
    const _Float16* th = thetaT + ((size_t)b * 4096 + qbase + mrow) * 32;
#pragma unroll
    for (int e = 0; e < 16; ++e) thA[e] = th[ka_idx(e, grp)];
  }

  const _Float16* phib = phiP + (size_t)b * 1024 * 32;
  const _Float16* gb   = gT   + (size_t)b * 128 * 1024;

  float rm[8];
#pragma unroll
  for (int r = 0; r < 8; ++r) rm[r] = -3.0e38f;

  // ---- pass 1: scores -> LDS, running per-lane row max -------------------
  for (int jb = 0; jb < 1024; jb += 32) {
    const _Float16* p0r = phib + (size_t)(jb + mrow) * 32;
    const _Float16* p1r = phib + (size_t)(jb + 16 + mrow) * 32;
    __builtin_prefetch(phib + (size_t)(jb + 32 + mrow) * 32, 0, 1);
    v16h b0, b1;
#pragma unroll
    for (int e = 0; e < 16; ++e) {
      int c = kb_idx(e, grp);
      b0[e] = p0r[c];
      b1[e] = p1r[c];
    }
    v8f s0 = __builtin_amdgcn_wmma_f32_16x16x32_f16(
        false, thA, false, b0, (short)0, zero8(), false, false);
    v8f s1 = __builtin_amdgcn_wmma_f32_16x16x32_f16(
        false, thA, false, b1, (short)0, zero8(), false, false);
#pragma unroll
    for (int r = 0; r < 8; ++r) {
      rm[r] = fmaxf(rm[r], fmaxf(s0[r], s1[r]));
      Sw[(r + 8 * grp) * SROW + jb + mrow]      = (_Float16)s0[r];
      Sw[(r + 8 * grp) * SROW + jb + 16 + mrow] = (_Float16)s1[r];
    }
  }
  // one cross-lane reduction for all 1024 keys
#pragma unroll
  for (int r = 0; r < 8; ++r)
#pragma unroll
    for (int mask = 1; mask < 16; mask <<= 1)
      rm[r] = fmaxf(rm[r], __shfl_xor(rm[r], mask, 32));
  if (mrow == 0) {  // lanes 0 (rows 0..7) and 16 (rows 8..15)
#pragma unroll
    for (int r = 0; r < 8; ++r) Rmax[r + 8 * grp] = rm[r];
  }

  // Distributed async copy of g block jb (128 rows x 64B) into buffer `buf`:
  // each wave issues 2 instructions covering rows wave*16 .. wave*16+15;
  // lane covers row (i*8 + lane/4), 16B segment (lane&3).
  auto issue_g = [&](int jb, int buf) {
    const int seg = lane & 3;
#pragma unroll
    for (int j = 0; j < 2; ++j) {
      const int row = (wave * 2 + j) * 8 + (lane >> 2);
      unsigned lds_off =
          (unsigned)(size_t)(Gbuf + ((size_t)buf * 128 + row) * GROW) +
          (unsigned)(seg * 16);
      unsigned long long ga =
          (unsigned long long)(size_t)(gb + (size_t)row * 1024 + jb) +
          (unsigned long long)(seg * 16);
      asm volatile("global_load_async_to_lds_b128 %0, %1, off"
                   :: "v"(lds_off), "v"(ga)
                   : "memory");
    }
  };

  issue_g(0, 0);
  __syncthreads();  // also publishes Rmax and pass-1 S tiles
  const float rowmax = Rmax[mrow];  // max of this lane's query row

  // ---- pass 2: P = exp(S - max) in A-layout registers, O += P @ g --------
  v8f acc[8];
#pragma unroll
  for (int t = 0; t < 8; ++t) acc[t] = zero8();
  float lsum = 0.0f;

  for (int kb = 0; kb < 32; ++kb) {
    const int jb = kb * 32;
    // prefetch next g block into the other buffer (wraps; always in-bounds)
    issue_g(((kb + 1) & 31) * 32, (kb + 1) & 1);
    // per-wave: own portion of current buffer complete; barrier covers rest
    asm volatile("s_wait_asynccnt 0x2" ::: "memory");
    __syncthreads();

    v16h pA;
#pragma unroll
    for (int e = 0; e < 16; ++e) {
      float v = (float)Sw[mrow * SROW + jb + ka_idx(e, grp)];
      float p = __expf(v - rowmax);
      lsum += p;
      pA[e] = (_Float16)p;
    }
    const _Float16* gl = Gbuf + (size_t)(kb & 1) * 128 * GROW;
#pragma unroll
    for (int t = 0; t < 8; ++t) {
      const _Float16* gr = gl + (size_t)(t * 16 + mrow) * GROW + 16 * grp;
      v16h bg;
#pragma unroll
      for (int e = 0; e < 16; ++e) bg[e] = gr[e];  // contiguous 32B, no conflicts
      acc[t] = __builtin_amdgcn_wmma_f32_16x16x32_f16(
          false, pA, false, bg, (short)0, acc[t], false, false);
    }
    __syncthreads();  // all waves done with this buffer before reuse
  }

  // full row sum: other half-wave holds the complementary key subset
  lsum += __shfl_xor(lsum, 16, 32);
  float linv[8];
#pragma unroll
  for (int r = 0; r < 8; ++r) linv[r] = 1.0f / __shfl(lsum, r + 8 * grp, 32);

  _Float16* agb = ag + ((size_t)b * 4096 + qbase) * 128;
#pragma unroll
  for (int r = 0; r < 8; ++r)
#pragma unroll
    for (int t = 0; t < 8; ++t)
      agb[(size_t)(r + 8 * grp) * 128 + t * 16 + mrow] =
          (_Float16)(acc[t][r] * linv[r]);
}

// ---------------------------------------------------------------------------
// Kernel 4: final 1x1 conv + residual (wf pre-packed: 16 tiles x 4 chunks).
// ---------------------------------------------------------------------------
__global__ void __launch_bounds__(256) final_kernel(
    const _Float16* __restrict__ ag, const _Float16* __restrict__ wfP,
    const float* __restrict__ x, const float* __restrict__ sigma_p,
    float* __restrict__ out) {
  const int b = blockIdx.y;
  const int lane = threadIdx.x & 31;
  const int wave = threadIdx.x >> 5;
  const int nbase = blockIdx.x * 16;
  const int mrow = lane & 15;
  const int grp  = lane >> 4;
  const float sigma = *sigma_p;

  const _Float16* agb = ag + ((size_t)b * 4096 + nbase + mrow) * 128;
  v8f acc0 = zero8(), acc1 = zero8();

  auto ldp = [&](int t, int k) {
    return *(const v16h*)(wfP + (((size_t)t * 4 + k) * 32 + lane) * 16);
  };

  for (int k = 0; k < 4; ++k) {
    v16h bop;
#pragma unroll
    for (int e = 0; e < 16; ++e) bop[e] = agb[k * 32 + kb_idx(e, grp)];
    acc0 = __builtin_amdgcn_wmma_f32_16x16x32_f16(
        false, ldp(wave, k), false, bop, (short)0, acc0, false, false);
    acc1 = __builtin_amdgcn_wmma_f32_16x16x32_f16(
        false, ldp(wave + 8, k), false, bop, (short)0, acc1, false, false);
  }

  const float* xb = x + (size_t)b * 256 * 4096;
  float* ob = out + (size_t)b * 256 * 4096;
  const int ncol = nbase + mrow;
#pragma unroll
  for (int r = 0; r < 8; ++r) {
    int o0 = wave * 16 + r + 8 * grp;
    int o1 = (wave + 8) * 16 + r + 8 * grp;
    ob[(size_t)o0 * 4096 + ncol] =
        sigma * acc0[r] + xb[(size_t)o0 * 4096 + ncol];
    ob[(size_t)o1 * 4096 + ncol] =
        sigma * acc1[r] + xb[(size_t)o1 * 4096 + ncol];
  }
}

// ---------------------------------------------------------------------------
extern "C" void kernel_launch(void* const* d_in, const int* in_sizes, int n_in,
                              void* d_out, int out_size, void* d_ws,
                              size_t ws_size, hipStream_t stream) {
  (void)in_sizes; (void)n_in; (void)out_size; (void)ws_size;
  const float* x       = (const float*)d_in[0];
  const float* w_theta = (const float*)d_in[1];
  const float* w_phi   = (const float*)d_in[2];
  const float* w_g     = (const float*)d_in[3];
  const float* w_final = (const float*)d_in[4];
  const float* sigma   = (const float*)d_in[5];
  float* out = (float*)d_out;

  // Workspace layout (f16 intermediates + packed weights, ~46 MB total):
  char* ws = (char*)d_ws;
  const size_t MB = 1024 * 1024;
  _Float16* thetaT = (_Float16*)(ws + 0 * MB);   // 16*4096*32  = 4 MB
  _Float16* phiF   = (_Float16*)(ws + 4 * MB);   // 16*4096*32  = 4 MB
  _Float16* gF     = (_Float16*)(ws + 8 * MB);   // 16*4096*128 = 16 MB
  _Float16* phiP   = (_Float16*)(ws + 24 * MB);  // 16*1024*32  = 1 MB
  _Float16* gT     = (_Float16*)(ws + 25 * MB);  // 16*128*1024 = 4 MB
  _Float16* ag     = (_Float16*)(ws + 29 * MB);  // 16*4096*128 = 16 MB
  _Float16* wtP    = (_Float16*)(ws + 45 * MB);            // 16 KB
  _Float16* wpP    = (_Float16*)(ws + 45 * MB + 32768);    // 16 KB
  _Float16* wgP    = (_Float16*)(ws + 45 * MB + 65536);    // 64 KB
  _Float16* wfP    = (_Float16*)(ws + 45 * MB + 131072);   // 64 KB

  dim3 blk(256);

  // 0) pre-pack weights into WMMA A-operand lane layout (f16)
  pack_w_kernel<<<(2 * 8 * 512 + 255) / 256, blk, 0, stream>>>(w_theta, wtP, 2, 8);
  pack_w_kernel<<<(2 * 8 * 512 + 255) / 256, blk, 0, stream>>>(w_phi, wpP, 2, 8);
  pack_w_kernel<<<(8 * 8 * 512 + 255) / 256, blk, 0, stream>>>(w_g, wgP, 8, 8);
  pack_w_kernel<<<(16 * 4 * 512 + 255) / 256, blk, 0, stream>>>(w_final, wfP, 16, 4);

  // 1) fused projections: one pass over x
  proj_fused_kernel<<<dim3(32, 16), blk, 0, stream>>>(
      x, wtP, wpP, wgP, thetaT, phiF, gF);

  // 2) 2x2 maxpool
  {
    int tot_phi = 16 * 1024 * 32;
    pool2_phi_kernel<<<(tot_phi + 255) / 256, blk, 0, stream>>>(phiF, phiP,
                                                                tot_phi);
    int tot_g = 16 * 1024 * 128;
    pool2_g_kernel<<<(tot_g + 255) / 256, blk, 0, stream>>>(gF, gT, tot_g);
  }

  // 3) attention: S tiles (258 KB) + Rmax + double-buffered async g (20 KB)
  size_t attn_lds = (size_t)8 * 16 * 1032 * 2 + 8 * 16 * 4 +
                    (size_t)2 * 128 * 40 * 2;  // ~278.5 KB
  attn_kernel<<<dim3(32, 16), blk, attn_lds, stream>>>(thetaT, phiP, gT, ag);

  // 4) final conv + residual (fully overwrites d_out)
  final_kernel<<<dim3(256, 16), blk, 0, stream>>>(ag, wfP, x, sigma, out);
}
